// SpatialAttention_63213328663162
// MI455X (gfx1250) — matrix operand order repaired and checked by reading, not compiled, not run on hardware
//
#include <hip/hip_runtime.h>

#define BB 16
#define DMODEL 512
#define HEADS 8
#define NN 512
#define LL 12
#define DK 64
#define PP (NN * LL) /* 6144 */

typedef __attribute__((ext_vector_type(16))) __bf16 bf16x16;
typedef __attribute__((ext_vector_type(8))) float f32x8;

union Frag {
    bf16x16 v;
    uint4 q[2];
};

__device__ __forceinline__ __bf16 f2bf(float f) {
    unsigned u = __builtin_bit_cast(unsigned, f);
    unsigned r = u + 0x7fffu + ((u >> 16) & 1u);
    unsigned short h = (unsigned short)(r >> 16);
    return __builtin_bit_cast(__bf16, h);
}
__device__ __forceinline__ float bf2f(__bf16 b) {
    unsigned short s = __builtin_bit_cast(unsigned short, b);
    unsigned u = ((unsigned)s) << 16;
    return __builtin_bit_cast(float, u);
}

// A fragment: 16(M) x 32(K) bf16 from row-major [M][K], rstride in elements.
// Lanes 0-15 row m hold K 0..7,16..23; lanes 16-31 hold K 8..15,24..31.
__device__ __forceinline__ void load_a(Frag& f, const __bf16* base, int rstride) {
    int lane = threadIdx.x & 31;
    int row = lane & 15;
    int hi = lane >> 4;
    const __bf16* p = base + (size_t)row * rstride + hi * 8;
    f.q[0] = *(const uint4*)(p);
    f.q[1] = *(const uint4*)(p + 16);
}
// B fragment: 32(K) x 16(N) bf16 loaded from B-transposed row-major [N][K].
// Lanes 0-15 col n hold K 0..7,16..23; lanes 16-31 hold K 8..15,24..31.
__device__ __forceinline__ void load_b(Frag& f, const __bf16* baseT, int rstride) {
    int lane = threadIdx.x & 31;
    int col = lane & 15;
    int hi = lane >> 4;
    const __bf16* p = baseT + (size_t)col * rstride + hi * 8;
    f.q[0] = *(const uint4*)(p);
    f.q[1] = *(const uint4*)(p + 16);
}

__device__ __forceinline__ f32x8 wmma_bf16(const Frag& a, const Frag& b, f32x8 c) {
    return __builtin_amdgcn_wmma_f32_16x16x32_bf16(false, a.v, false, b.v, (short)0, c,
                                                   false, false);
}

// ---------------- prep: weights -> bf16, wsum = sum(weights_pool) ----------------
__global__ __launch_bounds__(256) void k_prep_w(const float* __restrict__ Wq,
                                                const float* __restrict__ Wv,
                                                const float* __restrict__ Wc,
                                                const float* __restrict__ wpool,
                                                __bf16* __restrict__ wq, __bf16* __restrict__ wv,
                                                __bf16* __restrict__ wc,
                                                float* __restrict__ wsum) {
    int i = blockIdx.x * 256 + threadIdx.x;
    if (i < DMODEL * DMODEL) {
        wq[i] = f2bf(Wq[i]);
        wv[i] = f2bf(Wv[i]);
        wc[i] = f2bf(Wc[i]);
    }
    if (i == 0) {
        float s = 0.f;
        for (int j = 0; j < LL * 9; ++j) s += wpool[j];
        *wsum = s;
    }
}

// ---------------- prep: Aapt = softmax(relu(nv1@nv2)); biasN = Aapt @ bias_pool ----------------
__global__ __launch_bounds__(64) void k_aapt_bias(const float* __restrict__ nv1,
                                                  const float* __restrict__ nv2,
                                                  const float* __restrict__ bias_pool,
                                                  float* __restrict__ biasN) {
    __shared__ float row[NN];
    __shared__ float red[64];
    int n = blockIdx.x, t = threadIdx.x;
    float a[10];
#pragma unroll
    for (int k = 0; k < 10; ++k) a[k] = nv1[n * 10 + k];
    float lmax = -1e30f;
    for (int m = t; m < NN; m += 64) {
        float s = 0.f;
#pragma unroll
        for (int k = 0; k < 10; ++k) s += a[k] * nv2[k * NN + m];
        s = fmaxf(s, 0.f);
        row[m] = s;
        lmax = fmaxf(lmax, s);
    }
    red[t] = lmax;
    __syncthreads();
    for (int s = 32; s > 0; s >>= 1) {
        if (t < s) red[t] = fmaxf(red[t], red[t + s]);
        __syncthreads();
    }
    float mx = red[0];
    __syncthreads();
    float lsum = 0.f;
    for (int m = t; m < NN; m += 64) {
        float e = __expf(row[m] - mx);
        row[m] = e;
        lsum += e;
    }
    red[t] = lsum;
    __syncthreads();
    for (int s = 32; s > 0; s >>= 1) {
        if (t < s) red[t] += red[t + s];
        __syncthreads();
    }
    float inv = 1.f / red[0];
    float s = 0.f;
    for (int m = 0; m < NN; ++m) s += row[m] * bias_pool[m * DK + t];
    biasN[n * DK + t] = s * inv;
}

// ---------------- prep: key softmax over dk, store transposed [h][l][dk][n] bf16 ----------------
__global__ __launch_bounds__(64) void k_key_softmax(const float* __restrict__ memory,
                                                    __bf16* __restrict__ keyT) {
    int n = blockIdx.x, l = blockIdx.y, h = blockIdx.z, t = threadIdx.x;
    __shared__ float red[64];
    size_t base = (((size_t)h * LL + l) * NN + n) * DK;
    float x = memory[base + t] * 0.125f;
    red[t] = x;
    __syncthreads();
    for (int s = 32; s > 0; s >>= 1) {
        if (t < s) red[t] = fmaxf(red[t], red[t + s]);
        __syncthreads();
    }
    float mx = red[0];
    __syncthreads();
    float e = __expf(x - mx);
    red[t] = e;
    __syncthreads();
    for (int s = 32; s > 0; s >>= 1) {
        if (t < s) red[t] += red[t + s];
        __syncthreads();
    }
    keyT[(((size_t)h * LL + l) * DK + t) * NN + n] = f2bf(e / red[0]);
}

// ---------------- prep: x [b][c][p] fp32 -> xT [b][p][c] bf16 (L2-resident B operand) -------------
__global__ __launch_bounds__(256) void k_x_transpose(const float* __restrict__ x,
                                                     __bf16* __restrict__ xT) {
    const int pb = blockIdx.x;  // 96 tiles of 64 p
    const int cb = blockIdx.y;  // 8 tiles of 64 c
    const int b = blockIdx.z;
    const int t = threadIdx.x;
    __shared__ __align__(16) __bf16 tile[64][72];
    const size_t xbase = (size_t)b * DMODEL * PP + (size_t)cb * 64 * PP + (size_t)pb * 64;
    for (int e = t; e < 64 * 64; e += 256) {
        int p = e & 63, c = e >> 6;
        tile[p][c] = f2bf(x[xbase + (size_t)c * PP + p]);
    }
    __syncthreads();
    __bf16* dst = xT + ((size_t)b * PP + (size_t)pb * 64) * DMODEL + cb * 64;
    for (int s = t; s < 64 * 8; s += 256) {
        int p = s >> 3, i = s & 7;
        *(uint4*)(dst + (size_t)p * DMODEL + i * 8) = *(const uint4*)&tile[p][i * 8];
    }
}

// ---------------- conv q/v: barrier-free WMMA GEMM; emit qT (softmaxed), vT, vN ----------------
__global__ __launch_bounds__(256) void k_conv_qv(const __bf16* __restrict__ xT,
                                                 const __bf16* __restrict__ wq,
                                                 const __bf16* __restrict__ wv,
                                                 const float* __restrict__ bq,
                                                 const float* __restrict__ bv,
                                                 __bf16* __restrict__ qT, __bf16* __restrict__ vT,
                                                 __bf16* __restrict__ vN) {
    const int pb = blockIdx.x;  // 64 blocks of 96 p (= 8 n x 12 l)
    const int ob = blockIdx.y;  // head block (MTILE = 64 = one head)
    const int b = blockIdx.z;
    const int t = threadIdx.x;
    const int w = t >> 5;
    const int wm = w >> 1;  // 0..3 -> 16-row M subtile
    const int wn = w & 1;   // 0..1 -> 48-col N subtile

    __shared__ float outS[64][100];

    const __bf16* abase_q = wq + ((size_t)ob * 64 + wm * 16) * DMODEL;
    const __bf16* abase_v = wv + ((size_t)ob * 64 + wm * 16) * DMODEL;
    const __bf16* bbase = xT + ((size_t)b * PP + (size_t)pb * 96 + wn * 48) * DMODEL;

    f32x8 accq[3] = {{0.f}}, accv[3] = {{0.f}};
    for (int kk = 0; kk < 16; ++kk) {
        const int c0 = kk * 32;
        Frag aq, av, bfr;
        load_a(aq, abase_q + c0, DMODEL);
        load_a(av, abase_v + c0, DMODEL);
#pragma unroll
        for (int j = 0; j < 3; ++j) {
            load_b(bfr, bbase + (size_t)(j * 16) * DMODEL + c0, DMODEL);
            accq[j] = wmma_bf16(aq, bfr, accq[j]);
            accv[j] = wmma_bf16(av, bfr, accv[j]);
        }
        if (kk + 1 < 16) {  // WGP-scope prefetch of next k-slice (high temporal locality)
            __builtin_prefetch(bbase + (size_t)((t & 31) & 15) * DMODEL + c0 + 32, 0, 3);
        }
    }
    const int lane = t & 31, colL = lane & 15, hi = lane >> 4;
    // ---- q: bias + relu -> LDS; softmax over dk -> qT[b,h,l,n,dk]
#pragma unroll
    for (int j = 0; j < 3; ++j)
#pragma unroll
        for (int r = 0; r < 8; ++r) {
            int m = wm * 16 + r + 8 * hi;
            int o = ob * 64 + m;
            outS[m][wn * 48 + j * 16 + colL] = fmaxf(accq[j][r] + bq[o], 0.f);
        }
    __syncthreads();
    if (t < 96) {
        int p = pb * 96 + t;
        int n = p / LL, l = p % LL;
        float mx = -1e30f;
        for (int m = 0; m < 64; ++m) mx = fmaxf(mx, outS[m][t] * 0.125f);
        float sm = 0.f;
        for (int m = 0; m < 64; ++m) sm += __expf(outS[m][t] * 0.125f - mx);
        float inv = 1.f / sm;
        uint4 pk[8];
        __bf16* pe = (__bf16*)pk;
        for (int m = 0; m < 64; ++m) pe[m] = f2bf(__expf(outS[m][t] * 0.125f - mx) * inv);
        uint4* dst = (uint4*)(qT + ((((size_t)b * HEADS + ob) * LL + l) * NN + n) * DK);
#pragma unroll
        for (int i = 0; i < 8; ++i) dst[i] = pk[i];
    }
    __syncthreads();
    // ---- v: bias + relu -> LDS; emit vN[b,h,l,n,dk] and vT[b,h,l,dk,n]
#pragma unroll
    for (int j = 0; j < 3; ++j)
#pragma unroll
        for (int r = 0; r < 8; ++r) {
            int m = wm * 16 + r + 8 * hi;
            int o = ob * 64 + m;
            outS[m][wn * 48 + j * 16 + colL] = fmaxf(accv[j][r] + bv[o], 0.f);
        }
    __syncthreads();
    if (t < 96) {
        int p = pb * 96 + t;
        int n = p / LL, l = p % LL;
        uint4 pk[8];
        __bf16* pe = (__bf16*)pk;
        for (int m = 0; m < 64; ++m) pe[m] = f2bf(outS[m][t]);
        uint4* dst = (uint4*)(vN + ((((size_t)b * HEADS + ob) * LL + l) * NN + n) * DK);
#pragma unroll
        for (int i = 0; i < 8; ++i) dst[i] = pk[i];
    }
    for (int s = t; s < 64 * LL; s += 256) {
        int dk = s / LL, l = s % LL;
        int n0 = pb * 8;
        uint4 pk;
        __bf16* pe = (__bf16*)&pk;
#pragma unroll
        for (int i = 0; i < 8; ++i) pe[i] = f2bf(outS[dk][i * LL + l]);
        *(uint4*)(vT + ((((size_t)b * HEADS + ob) * LL + l) * DK + dk) * NN + n0) = pk;
    }
}

// ---------------- kv[x][y] = sum_n keyT[x][n] * v[n][y]; emit kvT[y][x] bf16 ----------------
__global__ __launch_bounds__(256) void k_kv(const __bf16* __restrict__ keyT,
                                            const __bf16* __restrict__ vT,
                                            __bf16* __restrict__ kvT) {
    const int l = blockIdx.x, h = blockIdx.y, b = blockIdx.z;
    const int t = threadIdx.x, w = t >> 5;
    const int wm = w >> 1, wn2 = w & 1;
    __shared__ float kvS[64][66];
    const __bf16* kp = keyT + ((size_t)h * LL + l) * DK * NN;
    const __bf16* vp = vT + (((size_t)b * HEADS + h) * LL + l) * DK * NN;
    f32x8 acc[2] = {{0.f}};
    for (int kk = 0; kk < 16; ++kk) {
        int n0 = kk * 32;
        Frag a, b0, b1;
        load_a(a, kp + (size_t)(wm * 16) * NN + n0, NN);
        load_b(b0, vp + (size_t)(wn2 * 32) * NN + n0, NN);
        load_b(b1, vp + (size_t)(wn2 * 32 + 16) * NN + n0, NN);
        acc[0] = wmma_bf16(a, b0, acc[0]);
        acc[1] = wmma_bf16(a, b1, acc[1]);
    }
    const int lane = t & 31, colL = lane & 15, hi = lane >> 4;
#pragma unroll
    for (int j = 0; j < 2; ++j)
#pragma unroll
        for (int r = 0; r < 8; ++r)
            kvS[wm * 16 + r + 8 * hi][wn2 * 32 + j * 16 + colL] = acc[j][r];
    __syncthreads();
    int y = t >> 2, xq = t & 3;
    uint4 pk[2];
    __bf16* pe = (__bf16*)pk;
#pragma unroll
    for (int i = 0; i < 16; ++i) pe[i] = f2bf(kvS[xq * 16 + i][y]);
    uint4* dst = (uint4*)(kvT + ((((size_t)b * HEADS + h) * LL + l) * DK + y) * DK + xq * 16);
    dst[0] = pk[0];
    dst[1] = pk[1];
}

// ---------------- attn = q_sm @ kv ; += S*value + biasN ; emit out5T[b][p][c] bf16 ----------------
__global__ __launch_bounds__(256) void k_attn(const __bf16* __restrict__ qT,
                                              const __bf16* __restrict__ kvT,
                                              const __bf16* __restrict__ vN,
                                              const float* __restrict__ biasN,
                                              const float* __restrict__ wsum,
                                              __bf16* __restrict__ out5T) {
    const int nb = blockIdx.x, h = blockIdx.y, b = blockIdx.z;
    const int t = threadIdx.x, w = t >> 5;
    const int mi = w & 1, ni = w >> 1;
    __shared__ __align__(16) __bf16 attnS[LL][32][72];
    const int lane = t & 31, colL = lane & 15, hi = lane >> 4;
    for (int l = 0; l < LL; ++l) {
        const __bf16* qp = qT + ((((size_t)b * HEADS + h) * LL + l) * NN + nb * 32 + mi * 16) * DK;
        const __bf16* kp = kvT + ((((size_t)b * HEADS + h) * LL + l) * DK + ni * 16) * DK;
        f32x8 acc = {0.f};
#pragma unroll
        for (int kk = 0; kk < 2; ++kk) {
            Frag a, bf;
            load_a(a, qp + kk * 32, DK);
            load_b(bf, kp + kk * 32, DK);
            acc = wmma_bf16(a, bf, acc);
        }
#pragma unroll
        for (int r = 0; r < 8; ++r)
            attnS[l][mi * 16 + r + 8 * hi][ni * 16 + colL] = f2bf(acc[r]);
    }
    __syncthreads();
    const float S = *wsum;
    for (int s = t; s < 32 * LL; s += 256) {
        int nl = s & 31, l = s >> 5;
        int n = nb * 32 + nl;
        const uint4* ap = (const uint4*)&attnS[l][nl][0];
        const uint4* vp = (const uint4*)(vN + ((((size_t)b * HEADS + h) * LL + l) * NN + n) * DK);
        const float4* bp = (const float4*)(biasN + (size_t)n * DK);
        uint4 outq[8];
        __bf16* oe = (__bf16*)outq;
#pragma unroll
        for (int i = 0; i < 8; ++i) {
            uint4 aq = ap[i], vq = vp[i];
            const __bf16* ae = (const __bf16*)&aq;
            const __bf16* ve = (const __bf16*)&vq;
            float4 c0 = bp[i * 2], c1 = bp[i * 2 + 1];
            float bb[8] = {c0.x, c0.y, c0.z, c0.w, c1.x, c1.y, c1.z, c1.w};
#pragma unroll
            for (int j = 0; j < 8; ++j)
                oe[i * 8 + j] = f2bf(bf2f(ae[j]) + S * bf2f(ve[j]) + bb[j]);
        }
        uint4* dst = (uint4*)(out5T + ((size_t)b * PP + (size_t)n * LL + l) * DMODEL + h * 64);
#pragma unroll
        for (int i = 0; i < 8; ++i) dst[i] = outq[i];
    }
}

// ---------------- final conv + affine: out = relu(Wc@out5 + bc)*(weight+1) + bias_param ------------
__global__ __launch_bounds__(256) void k_conv_out(const __bf16* __restrict__ wc,
                                                  const float* __restrict__ bc,
                                                  const __bf16* __restrict__ out5T,
                                                  const float* __restrict__ weight,
                                                  const float* __restrict__ bias_param,
                                                  float* __restrict__ out) {
    const int pb = blockIdx.x;  // 96 blocks of 64 p
    const int ob = blockIdx.y;  // 4 blocks of 128 o
    const int b = blockIdx.z;
    const int t = threadIdx.x, w = t >> 5;
    const int lane = t & 31, colL = lane & 15, hi = lane >> 4;
    f32x8 acc[4] = {{0.f}};
    const __bf16* ap0 = wc + ((size_t)ob * 128 + w * 16) * DMODEL;
    const __bf16* bp0 = out5T + ((size_t)b * PP + pb * 64) * DMODEL;
    for (int kk = 0; kk < 16; ++kk) {
        int c0 = kk * 32;
        Frag a;
        load_a(a, ap0 + c0, DMODEL);
#pragma unroll
        for (int j = 0; j < 4; ++j) {
            Frag bf;
            load_b(bf, bp0 + (size_t)(j * 16) * DMODEL + c0, DMODEL);
            acc[j] = wmma_bf16(a, bf, acc[j]);
        }
        if (kk + 1 < 16) {
            __builtin_prefetch(bp0 + (size_t)((t & 31) & 15) * DMODEL + c0 + 32, 0, 3);
        }
    }
#pragma unroll
    for (int j = 0; j < 4; ++j)
#pragma unroll
        for (int r = 0; r < 8; ++r) {
            int o = ob * 128 + w * 16 + r + 8 * hi;
            int p = pb * 64 + j * 16 + colL;
            float vv = fmaxf(acc[j][r] + bc[o], 0.f);
            size_t wp = (size_t)o * PP + p;
            out[(size_t)b * DMODEL * PP + wp] = vv * (weight[wp] + 1.f) + bias_param[wp];
        }
}

extern "C" void kernel_launch(void* const* d_in, const int* in_sizes, int n_in, void* d_out,
                              int out_size, void* d_ws, size_t ws_size, hipStream_t stream) {
    (void)in_sizes; (void)n_in; (void)out_size; (void)ws_size;
    const float* x = (const float*)d_in[0];
    const float* Wq = (const float*)d_in[1];
    const float* bq = (const float*)d_in[2];
    const float* Wv = (const float*)d_in[3];
    const float* bv = (const float*)d_in[4];
    const float* Wc = (const float*)d_in[5];
    const float* bc = (const float*)d_in[6];
    const float* memory = (const float*)d_in[7];
    const float* weight = (const float*)d_in[8];
    const float* bias_param = (const float*)d_in[9];
    const float* nv1 = (const float*)d_in[10];
    const float* nv2 = (const float*)d_in[11];
    const float* wpool = (const float*)d_in[12];
    const float* bias_pool = (const float*)d_in[13];
    float* out = (float*)d_out;

    char* ws = (char*)d_ws;
    size_t off = 0;
    auto carve = [&](size_t bytes) -> char* {
        char* p = ws + off;
        off += (bytes + 255) & ~(size_t)255;
        return p;
    };
    const size_t WSZ = (size_t)DMODEL * DMODEL;             // 262144
    const size_t QSZ = (size_t)BB * HEADS * LL * NN * DK;   // 50,331,648
    const size_t KSZ = (size_t)HEADS * LL * DK * NN;        // 3,145,728
    const size_t KVSZ = (size_t)BB * HEADS * LL * DK * DK;  // 6,291,456
    const size_t O5SZ = (size_t)BB * PP * DMODEL;           // 50,331,648

    __bf16* wq_bf = (__bf16*)carve(WSZ * 2);
    __bf16* wv_bf = (__bf16*)carve(WSZ * 2);
    __bf16* wc_bf = (__bf16*)carve(WSZ * 2);
    __bf16* keyT = (__bf16*)carve(KSZ * 2);
    __bf16* xT = (__bf16*)carve(O5SZ * 2);
    __bf16* qT = (__bf16*)carve(QSZ * 2);
    __bf16* vT = (__bf16*)carve(QSZ * 2);
    __bf16* vN = (__bf16*)carve(QSZ * 2);
    __bf16* kvT = (__bf16*)carve(KVSZ * 2);
    __bf16* out5T = (__bf16*)carve(O5SZ * 2);
    float* biasN = (float*)carve((size_t)NN * DK * 4);
    float* wsum = (float*)carve(256);

    k_prep_w<<<(DMODEL * DMODEL + 255) / 256, 256, 0, stream>>>(Wq, Wv, Wc, wpool, wq_bf, wv_bf,
                                                                wc_bf, wsum);
    k_aapt_bias<<<NN, 64, 0, stream>>>(nv1, nv2, bias_pool, biasN);
    k_key_softmax<<<dim3(NN, LL, HEADS), 64, 0, stream>>>(memory, keyT);
    k_x_transpose<<<dim3(96, 8, BB), 256, 0, stream>>>(x, xT);
    k_conv_qv<<<dim3(64, HEADS, BB), 256, 0, stream>>>(xT, wq_bf, wv_bf, bq, bv, qT, vT, vN);
    k_kv<<<dim3(LL, HEADS, BB), 256, 0, stream>>>(keyT, vT, kvT);
    k_attn<<<dim3(16, HEADS, BB), 256, 0, stream>>>(qT, kvT, vN, biasN, wsum, out5T);
    k_conv_out<<<dim3(96, 4, BB), 256, 0, stream>>>(wc_bf, bc, out5T, weight, bias_param, out);

    // tuple tail: weight, bias_param copied verbatim
    const size_t mainElems = (size_t)BB * DMODEL * PP;
    const size_t affElems = (size_t)DMODEL * PP;
    hipMemcpyAsync(out + mainElems, (void*)weight, affElems * sizeof(float),
                   hipMemcpyDeviceToDevice, stream);
    hipMemcpyAsync(out + mainElems + affElems, (void*)bias_param, affElems * sizeof(float),
                   hipMemcpyDeviceToDevice, stream);
}
